// CrossAttention_52063593562208
// MI455X (gfx1250) — matrix operand order repaired and checked by reading, not compiled, not run on hardware
//
#include <hip/hip_runtime.h>

// ---------------------------------------------------------------------------
// CDNA5 (gfx1250) cross-attention: bf16 WMMA everywhere, f32 accumulate.
//   B=4, Q=512, KV=2048, HIDDEN=1024, HEADS=16, HEAD_DIM=64
// Pipeline: cvt->bf16, WMMA GEMM projections (V stored transposed),
// flash attention (S^T trick, log2-domain online softmax, LDS P repack),
// WMMA GEMM output projection + bias -> f32.
// ---------------------------------------------------------------------------

typedef __attribute__((ext_vector_type(16))) __bf16 v16bf;
typedef __attribute__((ext_vector_type(8)))  __bf16 v8bf;
typedef __attribute__((ext_vector_type(8)))  float  v8f;

#define BATCH   4
#define QLEN    512
#define KVLEN   2048
#define HID     1024
#define HEADS   16
#define HDIM    64

#define NEG_BIG (-1.0e30f)
#define LOG2E   1.4426950408889634f

// ---------------------------------------------------------------------------
// WMMA helper: D = A(16x32 bf16) * B(32x16 bf16) + C(16x16 f32)
// ---------------------------------------------------------------------------
__device__ __forceinline__ v8f wmma_bf16(v16bf a, v16bf b, v8f c) {
    return __builtin_amdgcn_wmma_f32_16x16x32_bf16(
        /*neg_a=*/false, a, /*neg_b=*/false, b,
        /*c_mod=*/(short)0, c, /*reuse_a=*/false, /*reuse_b=*/false);
}

// A-fragment (16x32, MxK): lane L: row = m0 + (L&15), half h = L>>4.
// Lane holds K = k0 + 16h .. +15: one contiguous 32B run.
__device__ __forceinline__ v16bf load_a_frag(const __bf16* __restrict__ base,
                                             int row_stride, int m0, int k0,
                                             int lane) {
    const int mr = lane & 15, h = lane >> 4;
    return *(const v16bf*)(base + (size_t)(m0 + mr) * row_stride + k0 + h * 16);
}

// B-fragment (32x16, KxN) sourced from W[N][K] row-major (K contiguous):
// lane L: col N = n0 + (L&15), holds K = k0 + 16h .. +15: one 32B run.
__device__ __forceinline__ v16bf load_b_frag(const __bf16* __restrict__ base,
                                             int row_stride, int n0, int k0,
                                             int lane) {
    const int nc = lane & 15, h = lane >> 4;
    return *(const v16bf*)(base + (size_t)(n0 + nc) * row_stride + k0 + h * 16);
}

// ---------------------------------------------------------------------------
// fp32 -> bf16 conversion (grid-stride)
// ---------------------------------------------------------------------------
__global__ void cvt_f32_bf16(const float* __restrict__ in,
                             __bf16* __restrict__ out, int n) {
    int i = blockIdx.x * blockDim.x + threadIdx.x;
    const int stride = gridDim.x * blockDim.x;
    for (; i < n; i += stride) out[i] = (__bf16)in[i];
}

// bool mask -> f32 additive bias (0 or -1e30); one v_fma applies it later.
__global__ void mask_to_bias(const unsigned char* __restrict__ mask,
                             float* __restrict__ bias, int n) {
    int i = blockIdx.x * blockDim.x + threadIdx.x;
    const int stride = gridDim.x * blockDim.x;
    for (; i < n; i += stride) bias[i] = mask[i] ? 0.0f : NEG_BIG;
}

// ---------------------------------------------------------------------------
// GEMM: C[M,N] = A[M,K] * W[N,K]^T  (nn.Linear semantics, both K-contiguous)
// MODE 0: store bf16 row-major [M][ldout]
// MODE 1: store bf16 transposed [N][ldout]   (ldout = total M; packed b128)
// MODE 2: store f32  row-major [M][ldout] + bias[n]
// Block: 256 threads = 8 waves as 2(M) x 4(N); wave tile 32M x 64N.
// ---------------------------------------------------------------------------
template <int MODE>
__global__ __launch_bounds__(256) void gemm_bf16(
    const __bf16* __restrict__ A, const __bf16* __restrict__ W,
    void* __restrict__ Cout, const float* __restrict__ bias,
    int M, int N, int K, int ldout) {

    const int tid  = threadIdx.x;
    const int lane = tid & 31;
    const int wv   = tid >> 5;
    const int nc   = lane & 15;
    const int hl   = lane >> 4;

    const int m0 = blockIdx.x * 64 + (wv >> 2) * 32;
    const int n0 = blockIdx.y * 256 + (wv & 3) * 64;

    v8f c[2][4];
#pragma unroll
    for (int i = 0; i < 2; ++i)
#pragma unroll
        for (int j = 0; j < 4; ++j) c[i][j] = (v8f)(0.0f);

    for (int k0 = 0; k0 < K; k0 += 32) {
        // stream the next weight chunk toward L2 (global_prefetch_b8)
        if (k0 + 32 < K)
            __builtin_prefetch(W + (size_t)(n0 + nc) * K + k0 + 32 + hl * 16, 0, 1);

        v16bf a0 = load_a_frag(A, K, m0,      k0, lane);
        v16bf a1 = load_a_frag(A, K, m0 + 16, k0, lane);
#pragma unroll
        for (int j = 0; j < 4; ++j) {
            v16bf b = load_b_frag(W, K, n0 + j * 16, k0, lane);
            c[0][j] = wmma_bf16(a0, b, c[0][j]);
            c[1][j] = wmma_bf16(a1, b, c[1][j]);
        }
    }

    // C layout: lane L -> N = n0' + (L&15); VGPR r -> M = m0' + r + 8*(L>>4)
    if constexpr (MODE == 0) {
        __bf16* Cb = (__bf16*)Cout;
#pragma unroll
        for (int i = 0; i < 2; ++i)
#pragma unroll
            for (int j = 0; j < 4; ++j)
#pragma unroll
                for (int r = 0; r < 8; ++r) {
                    const int row = m0 + i * 16 + r + 8 * hl;
                    const int col = n0 + j * 16 + nc;
                    Cb[(size_t)row * ldout + col] = (__bf16)c[i][j][r];
                }
    } else if constexpr (MODE == 1) {
        // Transposed store: output row = N, contiguous along M -> packed b128
        __bf16* Cb = (__bf16*)Cout;
#pragma unroll
        for (int i = 0; i < 2; ++i)
#pragma unroll
            for (int j = 0; j < 4; ++j) {
                v8bf pk;
#pragma unroll
                for (int r = 0; r < 8; ++r) pk[r] = (__bf16)c[i][j][r];
                const int nrow = n0 + j * 16 + nc;
                const int mcol = m0 + i * 16 + 8 * hl;
                *(v8bf*)(Cb + (size_t)nrow * ldout + mcol) = pk;
            }
    } else {
        float* Cf = (float*)Cout;
#pragma unroll
        for (int i = 0; i < 2; ++i)
#pragma unroll
            for (int j = 0; j < 4; ++j) {
                const int col = n0 + j * 16 + nc;
                const float bv = bias ? bias[col] : 0.0f;
#pragma unroll
                for (int r = 0; r < 8; ++r) {
                    const int row = m0 + i * 16 + r + 8 * hl;
                    Cf[(size_t)row * ldout + col] = c[i][j][r] + bv;
                }
            }
    }
}

// ---------------------------------------------------------------------------
// Flash attention (online softmax, log2 domain), one wave per
// (b, head, 16-row q tile). Computes S^T = K_tile * Q^T so the softmax row
// (over kv) lives in-lane: chunk reduction = 8 regs + one shfl_xor(16).
// Masking is one v_fma per element against a precomputed f32 bias.
// P^T repacked to B-operand layout via 1KB/wave LDS (intra-wave ds dep).
// O^T = V^T * P^T accumulated over kv chunks of 32.
// ---------------------------------------------------------------------------
__global__ __launch_bounds__(256) void flash_attn(
    const __bf16* __restrict__ qp,    // [B*Q][HID]  (h*64+d columns)
    const __bf16* __restrict__ kp,    // [B*KV][HID]
    const __bf16* __restrict__ vT,    // [HID][B*KV] (transposed V projection)
    const float* __restrict__ mbias,  // [B*KV] 0 or -1e30
    __bf16* __restrict__ attn_out) {  // [B*Q][HID]

    const int tid  = threadIdx.x;
    const int lane = tid & 31;
    const int wv   = tid >> 5;
    const int nc   = lane & 15;
    const int hl   = lane >> 4;

    const int gw    = blockIdx.x * 8 + wv;        // 2048 waves total
    const int qtile = gw & 31;                    // 32 q-tiles of 16
    const int hh    = (gw >> 5) & 15;             // head
    const int b     = gw >> 9;                    // batch
    const int q0    = qtile * 16;

    __shared__ __bf16 Pb[8][16 * 32];
    __bf16* Pbuf = &Pb[wv][0];

    // Q^T B-operand fragments (reused for all kv chunks): K-dim = d = 64
    const v16bf bq0 = load_b_frag(qp, HID, b * QLEN + q0, hh * HDIM,      lane);
    const v16bf bq1 = load_b_frag(qp, HID, b * QLEN + q0, hh * HDIM + 32, lane);

    v8f o[4];
#pragma unroll
    for (int t = 0; t < 4; ++t) o[t] = (v8f)(0.0f);
    float m_run = NEG_BIG, l_run = 0.0f;

    // fold 1/sqrt(64) and log2(e) into one FMA scale (log2-domain softmax)
    const float scale2 = 0.125f * LOG2E;

    for (int kc = 0; kc < KVLEN; kc += 32) {
        const int krow = b * KVLEN + kc;

        // prefetch next chunk of K rows toward L2
        if (kc + 32 < KVLEN)
            __builtin_prefetch(kp + (size_t)(krow + 32 + nc) * HID + hh * HDIM + hl * 32,
                               0, 1);

        // ---- S^T tiles: rows M = kv, cols N = q ----
        v16bf ak0a = load_a_frag(kp, HID, krow,      hh * HDIM,      lane);
        v16bf ak0b = load_a_frag(kp, HID, krow,      hh * HDIM + 32, lane);
        v16bf ak1a = load_a_frag(kp, HID, krow + 16, hh * HDIM,      lane);
        v16bf ak1b = load_a_frag(kp, HID, krow + 16, hh * HDIM + 32, lane);

        v8f s0 = (v8f)(0.0f), s1 = (v8f)(0.0f);
        s0 = wmma_bf16(ak0a, bq0, s0);
        s0 = wmma_bf16(ak0b, bq1, s0);
        s1 = wmma_bf16(ak1a, bq0, s1);
        s1 = wmma_bf16(ak1b, bq1, s1);

        // ---- scale + mask via one FMA each; lane's kv = kc+8hl+r (+16) ----
        const v8f bias0 = *(const v8f*)(mbias + (size_t)b * KVLEN + kc + 8 * hl);
        const v8f bias1 = *(const v8f*)(mbias + (size_t)b * KVLEN + kc + 16 + 8 * hl);

        float cmax = NEG_BIG;
#pragma unroll
        for (int r = 0; r < 8; ++r) {
            s0[r] = fmaf(s0[r], scale2, bias0[r]);
            s1[r] = fmaf(s1[r], scale2, bias1[r]);
            cmax = fmaxf(cmax, fmaxf(s0[r], s1[r]));
        }
        cmax = fmaxf(cmax, __shfl_xor(cmax, 16));

        const float m_new = fmaxf(m_run, cmax);
        const float alpha = exp2f(m_run - m_new);   // bare v_exp_f32

        v8bf pb0, pb1;
        float csum = 0.0f;
#pragma unroll
        for (int r = 0; r < 8; ++r) {
            const float p0 = exp2f(s0[r] - m_new);
            const float p1 = exp2f(s1[r] - m_new);
            csum += p0 + p1;
            pb0[r] = (__bf16)p0;
            pb1[r] = (__bf16)p1;
        }
        csum += __shfl_xor(csum, 16);
        l_run = l_run * alpha + csum;
        m_run = m_new;

        // rescale running O^T accumulators (lane's q = nc matches stats)
#pragma unroll
        for (int t = 0; t < 4; ++t)
#pragma unroll
            for (int r = 0; r < 8; ++r) o[t][r] *= alpha;

        // ---- P^T repack via LDS: Pbuf[q][kv], rows of 32 bf16 ----
        *(v8bf*)(Pbuf + nc * 32 + 8 * hl)      = pb0;  // kv 8hl..8hl+7
        *(v8bf*)(Pbuf + nc * 32 + 16 + 8 * hl) = pb1;  // kv 16+8hl..+7
        // same-wave ds store->load; compiler inserts s_wait_dscnt

        // B-operand P^T: lane reads Pbuf[nc][16*hl .. +15] (contiguous 32B)
        const v16bf pB = *(const v16bf*)(Pbuf + nc * 32 + 16 * hl);

        // ---- O^T += V^T_tile * P^T (K = kv chunk of 32) ----
#pragma unroll
        for (int t = 0; t < 4; ++t) {
            v16bf av = load_a_frag(vT, BATCH * KVLEN, hh * HDIM + t * 16,
                                   b * KVLEN + kc, lane);
            o[t] = wmma_bf16(av, pB, o[t]);
        }
    }

    // ---- normalize and write: O^T lane -> q = nc, d = t*16 + 8hl + r ----
    const float inv_l = 1.0f / l_run;
    const int row = b * QLEN + q0 + nc;
#pragma unroll
    for (int t = 0; t < 4; ++t) {
        v8bf ot;
#pragma unroll
        for (int r = 0; r < 8; ++r) ot[r] = (__bf16)(o[t][r] * inv_l);
        *(v8bf*)(attn_out + (size_t)row * HID + hh * HDIM + t * 16 + 8 * hl) = ot;
    }
}

// ---------------------------------------------------------------------------
// Host-side orchestration
// ---------------------------------------------------------------------------
extern "C" void kernel_launch(void* const* d_in, const int* in_sizes, int n_in,
                              void* d_out, int out_size, void* d_ws, size_t ws_size,
                              hipStream_t stream) {
    const float* query = (const float*)d_in[0];
    const float* keyv  = (const float*)d_in[1];
    const unsigned char* mask = (const unsigned char*)d_in[2];
    const float* Wq = (const float*)d_in[3];
    const float* Wk = (const float*)d_in[4];
    const float* Wv = (const float*)d_in[5];
    const float* Wo = (const float*)d_in[6];
    const float* bo = (const float*)d_in[7];
    float* out = (float*)d_out;

    char* ws = (char*)d_ws;
    size_t off = 0;
    auto carve = [&](size_t bytes) -> char* {
        char* p = ws + off;
        off += (bytes + 255) & ~(size_t)255;
        return p;
    };
    const size_t MQ = (size_t)BATCH * QLEN;    // 2048
    const size_t MK = (size_t)BATCH * KVLEN;   // 8192

    __bf16* qin  = (__bf16*)carve(MQ * HID * 2);
    __bf16* kvin = (__bf16*)carve(MK * HID * 2);
    __bf16* wqb  = (__bf16*)carve((size_t)HID * HID * 2);
    __bf16* wkb  = (__bf16*)carve((size_t)HID * HID * 2);
    __bf16* wvb  = (__bf16*)carve((size_t)HID * HID * 2);
    __bf16* wob  = (__bf16*)carve((size_t)HID * HID * 2);
    __bf16* qpj  = (__bf16*)carve(MQ * HID * 2);
    __bf16* kpj  = (__bf16*)carve(MK * HID * 2);
    __bf16* vTt  = (__bf16*)carve(MK * HID * 2);   // [HID][B*KV]
    __bf16* aout = (__bf16*)carve(MQ * HID * 2);
    float*  mbia = (float*)carve(MK * 4);          // mask bias

    // 1) fp32 -> bf16 + mask bias
    cvt_f32_bf16<<<1024, 256, 0, stream>>>(query, qin,  (int)(MQ * HID));
    cvt_f32_bf16<<<2048, 256, 0, stream>>>(keyv,  kvin, (int)(MK * HID));
    cvt_f32_bf16<<<512,  256, 0, stream>>>(Wq, wqb, HID * HID);
    cvt_f32_bf16<<<512,  256, 0, stream>>>(Wk, wkb, HID * HID);
    cvt_f32_bf16<<<512,  256, 0, stream>>>(Wv, wvb, HID * HID);
    cvt_f32_bf16<<<512,  256, 0, stream>>>(Wo, wob, HID * HID);
    mask_to_bias<<<32,   256, 0, stream>>>(mask, mbia, (int)MK);

    // 2) projections (WMMA GEMMs)
    gemm_bf16<0><<<dim3(MQ / 64, HID / 256), 256, 0, stream>>>(
        qin, wqb, qpj, nullptr, (int)MQ, HID, HID, HID);
    gemm_bf16<0><<<dim3(MK / 64, HID / 256), 256, 0, stream>>>(
        kvin, wkb, kpj, nullptr, (int)MK, HID, HID, HID);
    gemm_bf16<1><<<dim3(MK / 64, HID / 256), 256, 0, stream>>>(
        kvin, wvb, vTt, nullptr, (int)MK, HID, HID, (int)MK);

    // 3) flash attention: 2048 waves, 8 per block
    flash_attn<<<256, 256, 0, stream>>>(qpj, kpj, vTt, mbia, aout);

    // 4) output projection + bias -> f32 d_out
    gemm_bf16<2><<<dim3(MQ / 64, HID / 256), 256, 0, stream>>>(
        aout, wob, out, bo, (int)MQ, HID, HID, HID);
}
